// MSWinAttention_49675591745639
// MI455X (gfx1250) — compile-verified, matrix operand order checked by reading, not run on hardware
//
#include <hip/hip_runtime.h>
#include <hip/hip_bf16.h>
#include <math.h>

typedef __attribute__((ext_vector_type(16))) _Float16 v16h;
typedef __attribute__((ext_vector_type(8)))  float    v8f;

// Load a v16h WMMA fragment in the CDNA5 16-bit A-layout (also used for B^T).
// `row` points at the start of this lane's 32-half K-row.
// Lane L (L<16): K = {0..7, 16..23}; lane L>=16: K = {8..15, 24..31}.
__device__ __forceinline__ v16h frag_ld(const _Float16* __restrict__ row, int lane) {
    const int base = (lane & 16) ? 8 : 0;
    v16h f;
#pragma unroll
    for (int j = 0; j < 8; ++j) {
        f[j]     = row[base + j];
        f[j + 8] = row[base + 16 + j];
    }
    return f;
}

__device__ __forceinline__ v8f wmma_f16(v16h a, v16h b, v8f c) {
    return __builtin_amdgcn_wmma_f32_16x16x32_f16(false, a, false, b, (short)0, c,
                                                  false, false);
}

// ---------------------------------------------------------------------------
// Kernel 1: LayerNorm over C=128 + Q projection (65536x128 @ 128x128) -> f16 q
// Block: 128 threads (4 waves), 64 token rows per block.
// ---------------------------------------------------------------------------
__global__ __launch_bounds__(128) void msw_ln_q_kernel(
    const float* __restrict__ inp, const float* __restrict__ nw,
    const float* __restrict__ nb, const float* __restrict__ qw,
    const float* __restrict__ qb, _Float16* __restrict__ q16) {
    __shared__ _Float16 sX[64 * 128];   // 16 KB  normalized activations (f16)
    __shared__ _Float16 sW[128 * 128];  // 32 KB  q_w (f16)
    __shared__ float    sS[128], sQ[128];

    const int t = threadIdx.x;
    const int row = t & 63, half = t >> 6;
    const int R0 = blockIdx.x * 64;
    const int g  = R0 + row;
    const int nidx = g >> 14;          // / 16384 (H*W)
    const int pos  = g & 16383;
    const float* src = inp + ((size_t)nidx * 128 + half * 64) * 16384 + pos;

    float xr[64];
    float s = 0.f, sq = 0.f;
#pragma unroll 8
    for (int i = 0; i < 64; ++i) {
        float v = src[(size_t)i * 16384];
        xr[i] = v; s += v; sq += v * v;
    }
    sS[t] = s; sQ[t] = sq;
    for (int i = t; i < 128 * 128; i += 128) sW[i] = (_Float16)qw[i];
    __syncthreads();

    float st  = sS[t ^ 64] + s;
    float sqt = sQ[t ^ 64] + sq;
    float mean = st * (1.f / 128.f);
    float var  = sqt * (1.f / 128.f) - mean * mean;
    float rs   = rsqrtf(var + 1e-5f);
    for (int i = 0; i < 64; ++i) {
        int c = half * 64 + i;
        sX[row * 128 + c] = (_Float16)((xr[i] - mean) * rs * nw[c] + nb[c]);
    }
    __syncthreads();

    const int wv = t >> 5, lane = t & 31;
    v8f zero = {0.f, 0.f, 0.f, 0.f, 0.f, 0.f, 0.f, 0.f};
    v8f acc[8];
#pragma unroll
    for (int nt = 0; nt < 8; ++nt) acc[nt] = zero;
#pragma unroll
    for (int ks = 0; ks < 4; ++ks) {
        v16h a = frag_ld(sX + (wv * 16 + (lane & 15)) * 128 + ks * 32, lane);
#pragma unroll
        for (int nt = 0; nt < 8; ++nt) {
            v16h b = frag_ld(sW + (nt * 16 + (lane & 15)) * 128 + ks * 32, lane);
            acc[nt] = wmma_f16(a, b, acc[nt]);
        }
    }
    const int rb = (lane & 16) ? 8 : 0;
    for (int nt = 0; nt < 8; ++nt) {
        int col = nt * 16 + (lane & 15);
        float bb = qb[col];
#pragma unroll
        for (int e = 0; e < 8; ++e) {
            int gr = R0 + wv * 16 + rb + e;
            q16[(size_t)gr * 128 + col] = (_Float16)(acc[nt][e] + bb);
        }
    }
}

// ---------------------------------------------------------------------------
// Kernel 2: patch-embed conv (stride==kernel) + LN + GELU + KV projection.
// One block per patch. Writes K rows (B,Lk,32) f16 and V^T (B,32,Lk) f16.
// ---------------------------------------------------------------------------
__global__ __launch_bounds__(256) void msw_conv_kv_kernel(
    const float* __restrict__ inp, const float* __restrict__ srw,
    const float* __restrict__ srb, const float* __restrict__ lnw,
    const float* __restrict__ lnb, const float* __restrict__ kvw,
    const float* __restrict__ kvb, _Float16* __restrict__ kbuf,
    _Float16* __restrict__ vtbuf, int n, int ksz) {
    __shared__ float sP[8192];   // im2col patch (K <= 128*8*8)
    __shared__ float sRed[256];
    __shared__ float sY[64];
    __shared__ float sG[64];
    __shared__ float sMV[2];

    const int b = blockIdx.x, p = blockIdx.y;
    const int Lk = gridDim.y;
    const int t = threadIdx.x;
    const int Hn = 128 / n, Wn = 128 / n, Pw = Wn / ksz;
    const int nidx = b / (n * n), rem = b % (n * n), wi = rem / n, wj = rem % n;
    const int pi = p / Pw, pj = p % Pw;
    const int bh = wi * Hn + pi * ksz, bw = wj * Wn + pj * ksz;
    const int kk = ksz * ksz, K = 128 * kk;

    for (int idx = t; idx < K; idx += 256) {
        int c = idx / kk, r2 = idx % kk, dy = r2 / ksz, dx = r2 % ksz;
        sP[idx] = inp[(((size_t)nidx * 128 + c) * 128 + bh + dy) * 128 + bw + dx];
    }
    __syncthreads();

    const int o = t >> 2, part = t & 3, Kq = K >> 2;
    const float* wrow = srw + (size_t)o * K + part * Kq;
    const float* prow = sP + part * Kq;
    float acc = 0.f;
    for (int i = 0; i < Kq; ++i) acc += prow[i] * wrow[i];
    sRed[t] = acc;
    __syncthreads();

    if (t < 64) {
        float y = sRed[t * 4] + sRed[t * 4 + 1] + sRed[t * 4 + 2] + sRed[t * 4 + 3] + srb[t];
        sY[t] = y;
    }
    __syncthreads();
    if (t == 0) {
        float m = 0.f, v = 0.f;
        for (int i = 0; i < 64; ++i) m += sY[i];
        m *= (1.f / 64.f);
        for (int i = 0; i < 64; ++i) { float d = sY[i] - m; v += d * d; }
        v *= (1.f / 64.f);
        sMV[0] = m; sMV[1] = rsqrtf(v + 1e-5f);
    }
    __syncthreads();
    if (t < 64) {
        float x = (sY[t] - sMV[0]) * sMV[1] * lnw[t] + lnb[t];
        sG[t] = 0.5f * x * (1.f + erff(x * 0.70710678118654752f));  // exact GELU
    }
    __syncthreads();
    if (t < 64) {
        const float* wr = kvw + t * 64;
        float a2 = kvb[t];
        for (int i = 0; i < 64; ++i) a2 += sG[i] * wr[i];
        if (t < 32) kbuf[((size_t)b * Lk + p) * 32 + t] = (_Float16)a2;
        else        vtbuf[((size_t)b * 32 + (t - 32)) * Lk + p] = (_Float16)a2;
    }
}

// ---------------------------------------------------------------------------
// Kernel 3: windowed attention, one head. Block: 128 threads (4 waves),
// one window x 64 query rows. QK^T: one WMMA per tile (K=32=hd).
// Softmax via wave32 __shfl_xor reductions. A.V via WMMA from LDS.
// ---------------------------------------------------------------------------
template <int LK>
__global__ __launch_bounds__(128) void msw_attn_kernel(
    const _Float16* __restrict__ q16, const _Float16* __restrict__ kbuf,
    const _Float16* __restrict__ vtbuf, _Float16* __restrict__ heads16,
    int n, int head, float scale) {
    __shared__ _Float16 sK[LK * 32];   // key rows
    __shared__ _Float16 sVT[32 * LK];  // V^T (channel-major)
    __shared__ _Float16 sA[64 * LK];   // f16 probabilities

    const int b = blockIdx.x, chunk = blockIdx.y;
    const int t = threadIdx.x, wv = t >> 5, lane = t & 31;
    const int Hn = 128 / n, Wn = 128 / n;
    const int nidx = b / (n * n), rem = b % (n * n), wi = rem / n, wj = rem % n;
    const int rowbase = nidx * 16384 + wi * Hn * 128 + wj * Wn;

    {
        const _Float16* gk = kbuf + (size_t)b * LK * 32;
        const _Float16* gv = vtbuf + (size_t)b * 32 * LK;
        for (int i = t; i < LK * 32; i += 128) { sK[i] = gk[i]; sVT[i] = gv[i]; }
    }
    __syncthreads();

    // Q fragment: gather this lane's query row (window -> global mapping)
    const int p0 = chunk * 64 + wv * 16 + (lane & 15);
    const int gq = rowbase + (p0 / Wn) * 128 + (p0 % Wn);
    v16h qf = frag_ld(q16 + (size_t)gq * 128 + head * 32, lane);

    constexpr int NT = LK / 16;
    v8f zero = {0.f, 0.f, 0.f, 0.f, 0.f, 0.f, 0.f, 0.f};
    float lg[NT][8];
#pragma unroll
    for (int nt = 0; nt < NT; ++nt) {
        v16h kf = frag_ld(sK + (nt * 16 + (lane & 15)) * 32, lane);
        v8f c = wmma_f16(qf, kf, zero);
#pragma unroll
        for (int e = 0; e < 8; ++e) lg[nt][e] = c[e] * scale;
    }

    // Row softmax: reduce over N (columns) = over 16 lanes of this half + tiles
    float rmax[8];
#pragma unroll
    for (int e = 0; e < 8; ++e) rmax[e] = lg[0][e];
#pragma unroll
    for (int nt = 1; nt < NT; ++nt)
#pragma unroll
        for (int e = 0; e < 8; ++e) rmax[e] = fmaxf(rmax[e], lg[nt][e]);
#pragma unroll
    for (int m = 1; m <= 8; m <<= 1)
#pragma unroll
        for (int e = 0; e < 8; ++e)
            rmax[e] = fmaxf(rmax[e], __shfl_xor(rmax[e], m, 32));
    float rsum[8];
#pragma unroll
    for (int e = 0; e < 8; ++e) rsum[e] = 0.f;
#pragma unroll
    for (int nt = 0; nt < NT; ++nt)
#pragma unroll
        for (int e = 0; e < 8; ++e) {
            float ex = expf(lg[nt][e] - rmax[e]);
            lg[nt][e] = ex; rsum[e] += ex;
        }
#pragma unroll
    for (int m = 1; m <= 8; m <<= 1)
#pragma unroll
        for (int e = 0; e < 8; ++e) rsum[e] += __shfl_xor(rsum[e], m, 32);

    const int rb = (lane & 16) ? 8 : 0;
#pragma unroll
    for (int nt = 0; nt < NT; ++nt)
#pragma unroll
        for (int e = 0; e < 8; ++e)
            sA[(wv * 16 + rb + e) * LK + nt * 16 + (lane & 15)] =
                (_Float16)(lg[nt][e] / rsum[e]);
    __syncthreads();

    // A.V : M=16 queries x N=32 channels, K=LK
#pragma unroll
    for (int nt2 = 0; nt2 < 2; ++nt2) {
        v8f acc = zero;
#pragma unroll
        for (int ks = 0; ks < LK / 32; ++ks) {
            v16h af = frag_ld(sA + (wv * 16 + (lane & 15)) * LK + ks * 32, lane);
            v16h bf = frag_ld(sVT + (nt2 * 16 + (lane & 15)) * LK + ks * 32, lane);
            acc = wmma_f16(af, bf, acc);
        }
        int col = head * 32 + nt2 * 16 + (lane & 15);
#pragma unroll
        for (int e = 0; e < 8; ++e) {
            int pm = chunk * 64 + wv * 16 + rb + e;
            int gr = rowbase + (pm / Wn) * 128 + (pm % Wn);   // window-reverse
            heads16[(size_t)gr * 128 + col] = (_Float16)acc[e];
        }
    }
}

// ---------------------------------------------------------------------------
// Kernel 4: output projection (65536x128 f16) @ (128x128) + bias -> fp32 out
// ---------------------------------------------------------------------------
__global__ __launch_bounds__(128) void msw_proj_kernel(
    const _Float16* __restrict__ heads16, const float* __restrict__ pw,
    const float* __restrict__ pb, float* __restrict__ out) {
    __shared__ _Float16 sW[128 * 128];
    const int t = threadIdx.x;
    const int R0 = blockIdx.x * 64;
    for (int i = t; i < 128 * 128; i += 128) sW[i] = (_Float16)pw[i];
    __syncthreads();

    const int wv = t >> 5, lane = t & 31;
    v8f zero = {0.f, 0.f, 0.f, 0.f, 0.f, 0.f, 0.f, 0.f};
    v8f acc[8];
#pragma unroll
    for (int nt = 0; nt < 8; ++nt) acc[nt] = zero;
    const _Float16* arow = heads16 + (size_t)(R0 + wv * 16 + (lane & 15)) * 128;
#pragma unroll
    for (int ks = 0; ks < 4; ++ks) {
        v16h a = frag_ld(arow + ks * 32, lane);
#pragma unroll
        for (int nt = 0; nt < 8; ++nt) {
            v16h b = frag_ld(sW + (nt * 16 + (lane & 15)) * 128 + ks * 32, lane);
            acc[nt] = wmma_f16(a, b, acc[nt]);
        }
    }
    const int rb = (lane & 16) ? 8 : 0;
    for (int nt = 0; nt < 8; ++nt) {
        int col = nt * 16 + (lane & 15);
        float bb = pb[col];
#pragma unroll
        for (int e = 0; e < 8; ++e) {
            int gr = R0 + wv * 16 + rb + e;
            out[(size_t)gr * 128 + col] = acc[nt][e] + bb;
        }
    }
}

// ---------------------------------------------------------------------------
extern "C" void kernel_launch(void* const* d_in, const int* in_sizes, int n_in,
                              void* d_out, int out_size, void* d_ws, size_t ws_size,
                              hipStream_t stream) {
    (void)in_sizes; (void)n_in; (void)out_size; (void)ws_size;

    const float* inp    = (const float*)d_in[0];
    const float* norm_w = (const float*)d_in[1];
    const float* norm_b = (const float*)d_in[2];
    const float* q_w    = (const float*)d_in[3];
    const float* q_b    = (const float*)d_in[4];
    const float* srw[4] = {(const float*)d_in[5],  (const float*)d_in[11],
                           (const float*)d_in[17], (const float*)d_in[23]};
    const float* srb[4] = {(const float*)d_in[6],  (const float*)d_in[12],
                           (const float*)d_in[18], (const float*)d_in[24]};
    const float* lnw[4] = {(const float*)d_in[7],  (const float*)d_in[13],
                           (const float*)d_in[19], (const float*)d_in[25]};
    const float* lnb[4] = {(const float*)d_in[8],  (const float*)d_in[14],
                           (const float*)d_in[20], (const float*)d_in[26]};
    const float* kvw[4] = {(const float*)d_in[9],  (const float*)d_in[15],
                           (const float*)d_in[21], (const float*)d_in[27]};
    const float* kvb[4] = {(const float*)d_in[10], (const float*)d_in[16],
                           (const float*)d_in[22], (const float*)d_in[28]};
    const float* proj_w = (const float*)d_in[29];
    const float* proj_b = (const float*)d_in[30];

    // Workspace layout (halves): q16 | heads16 | K rows | V^T   (~35 MB total)
    _Float16* q16     = (_Float16*)d_ws;
    _Float16* heads16 = q16 + 8388608;        // 65536*128
    _Float16* kb      = heads16 + 8388608;
    _Float16* vb      = kb + 819200;
    const size_t koff[4] = {0, 32768, 163840, 294912};

    const int nwin[4] = {4, 16, 64, 256};
    const int lk[4]   = {256, 256, 64, 64};
    const int nn[4]   = {1, 2, 4, 8};
    const int ksz[4]  = {8, 4, 4, 2};
    const float scale = 0.17677669529663687f;  // 32^-0.5

    msw_ln_q_kernel<<<1024, 128, 0, stream>>>(inp, norm_w, norm_b, q_w, q_b, q16);

    for (int i = 0; i < 4; ++i)
        msw_conv_kv_kernel<<<dim3(nwin[i], lk[i]), 256, 0, stream>>>(
            inp, srw[i], srb[i], lnw[i], lnb[i], kvw[i], kvb[i],
            kb + koff[i], vb + koff[i], nn[i], ksz[i]);

    msw_attn_kernel<256><<<dim3(4, 256), 128, 0, stream>>>(
        q16, kb + koff[0], vb + koff[0], heads16, 1, 0, scale);
    msw_attn_kernel<256><<<dim3(16, 64), 128, 0, stream>>>(
        q16, kb + koff[1], vb + koff[1], heads16, 2, 1, scale);
    msw_attn_kernel<64><<<dim3(64, 16), 128, 0, stream>>>(
        q16, kb + koff[2], vb + koff[2], heads16, 4, 2, scale);
    msw_attn_kernel<64><<<dim3(256, 4), 128, 0, stream>>>(
        q16, kb + koff[3], vb + koff[3], heads16, 8, 3, scale);

    msw_proj_kernel<<<1024, 128, 0, stream>>>(heads16, proj_w, proj_b, (float*)d_out);
}
